// MoDBlock_40192303956221
// MI455X (gfx1250) — compile-verified, matrix operand order checked by reading, not run on hardware
//
#include <hip/hip_runtime.h>

// ---------------------------------------------------------------------------
// MoD transformer block for MI455X (gfx1250, wave32, WMMA).
// GEMMs: v_wmma_f32_16x16x32_bf16, LDS double-buffered via
// global_load_async_to_lds_b128 (ASYNCcnt) -- all B operands pre-transposed
// to N-major so A and B staging are identical async b128 streams.
// ---------------------------------------------------------------------------

#define BB 4
#define TT 4096
#define DD 2048
#define HH 16
#define HD 128
#define KS 512            // selected tokens per batch
#define MM (BB * KS)      // 2048 total selected rows
#define QKVN (3 * DD)     // 6144
#define DFF 5461
#define DFFP 5504         // padded to 128*43
#define EPSF 1e-6f

typedef __attribute__((ext_vector_type(16))) __bf16 v16bf;
typedef __attribute__((ext_vector_type(8)))  float  v8f;

union FragU { uint4 u[2]; v16bf v; __bf16 h[16]; };

__device__ __forceinline__ v16bf ld_frag(const __bf16* rowp, int half) {
  FragU f;
  f.u[0] = *(const uint4*)(rowp + (half << 3));
  f.u[1] = *(const uint4*)(rowp + 16 + (half << 3));
  return f.v;
}

// Async 16-byte global -> LDS copy (per-lane), tracked by ASYNCcnt.
__device__ __forceinline__ void async_b128(void* lds, const void* gp) {
  asm volatile("global_load_async_to_lds_b128 %0, %1, off"
               :: "v"((unsigned)(uintptr_t)lds), "v"(gp)
               : "memory");
}

// ---------------------------------------------------------------------------
// Router scores: scores[b*T+t] = dot(x[b,t,:], router_w)
// ---------------------------------------------------------------------------
__global__ __launch_bounds__(256) void k_router(const float* __restrict__ x,
                                                const float* __restrict__ rw,
                                                float* __restrict__ scores) {
  long row = blockIdx.x;
  const float* xp = x + row * DD;
  float s = 0.f;
  for (int c = threadIdx.x; c < DD; c += 256) s += xp[c] * rw[c];
  __shared__ float red[256];
  red[threadIdx.x] = s;
  __syncthreads();
  for (int o = 128; o > 0; o >>= 1) {
    if (threadIdx.x < o) red[threadIdx.x] += red[threadIdx.x + o];
    __syncthreads();
  }
  if (threadIdx.x == 0) scores[row] = red[0];
}

// ---------------------------------------------------------------------------
// Per-batch top-K via LDS bitonic sort (ascending; take last K), then sort
// winning indices ascending. Tie-break matches reference: score + t*1e-6.
// ---------------------------------------------------------------------------
__global__ __launch_bounds__(1024) void k_topk(const float* __restrict__ scores,
                                               const int* __restrict__ position_ids,
                                               int* __restrict__ idx_out,
                                               int* __restrict__ pos_out) {
  __shared__ float key[TT];
  __shared__ int   id[TT];
  __shared__ int   sidx[KS];
  int b = blockIdx.x, tid = threadIdx.x;
  for (int i = tid; i < TT; i += 1024) {
    key[i] = scores[(long)b * TT + i] + (float)i * 1e-6f;
    id[i] = i;
  }
  __syncthreads();
  for (int k = 2; k <= TT; k <<= 1) {
    for (int j = k >> 1; j > 0; j >>= 1) {
      for (int i = tid; i < TT; i += 1024) {
        int ixj = i ^ j;
        if (ixj > i) {
          bool up = ((i & k) == 0);
          bool sw = up ? (key[i] > key[ixj]) : (key[i] < key[ixj]);
          if (sw) {
            float tk = key[i]; key[i] = key[ixj]; key[ixj] = tk;
            int   ti = id[i];  id[i]  = id[ixj];  id[ixj]  = ti;
          }
        }
      }
      __syncthreads();
    }
  }
  for (int i = tid; i < KS; i += 1024) sidx[i] = id[TT - KS + i];
  __syncthreads();
  for (int k = 2; k <= KS; k <<= 1) {
    for (int j = k >> 1; j > 0; j >>= 1) {
      for (int i = tid; i < KS; i += 1024) {
        int ixj = i ^ j;
        if (ixj > i) {
          bool up = ((i & k) == 0);
          bool sw = up ? (sidx[i] > sidx[ixj]) : (sidx[i] < sidx[ixj]);
          if (sw) { int t = sidx[i]; sidx[i] = sidx[ixj]; sidx[ixj] = t; }
        }
      }
      __syncthreads();
    }
  }
  for (int i = tid; i < KS; i += 1024) {
    int t = sidx[i];
    idx_out[b * KS + i] = t;
    pos_out[b * KS + i] = position_ids[(long)b * TT + t];
  }
}

// ---------------------------------------------------------------------------
// Transposing fp32 -> bf16 weight conversion with zero padding.
// src: (src_rows=K x src_cols=N) row-major fp32.
// dst: (NT x KT) row-major bf16, dst[n][k] = src[k][n], zero-padded.
// Grid: (KT/32, NT/32), 256 threads, 32x32 LDS tile.
// ---------------------------------------------------------------------------
__global__ __launch_bounds__(256) void k_cvt_padT(const float* __restrict__ src,
                                                  __bf16* __restrict__ dst,
                                                  int src_rows, int src_cols,
                                                  int NT, int KT) {
  __shared__ float tile[32][33];
  int k0 = blockIdx.x << 5, n0 = blockIdx.y << 5;
  int tx = threadIdx.x & 31, ty = threadIdx.x >> 5;  // ty in 0..7
#pragma unroll
  for (int i = ty; i < 32; i += 8) {
    int k = k0 + i, n = n0 + tx;
    tile[i][tx] = (k < src_rows && n < src_cols) ? src[(long)k * src_cols + n] : 0.f;
  }
  __syncthreads();
#pragma unroll
  for (int i = ty; i < 32; i += 8) {
    int n = n0 + i, k = k0 + tx;
    dst[(long)n * KT + k] = (__bf16)tile[tx][i];
  }
}

// ---------------------------------------------------------------------------
// V transpose: vt[z][n][k] = qkv[(b*KS+k)][2*DD + h*HD + n], z = b*H + h.
// Grid: (KS/32, HD/32, B*H), 256 threads.
// ---------------------------------------------------------------------------
__global__ __launch_bounds__(256) void k_transpose_v(const __bf16* __restrict__ qkv,
                                                     __bf16* __restrict__ vt) {
  __shared__ __bf16 t[32][33];
  int z = blockIdx.z, b = z >> 4, h = z & 15;
  int kk0 = blockIdx.x << 5, n0 = blockIdx.y << 5;
  int tx = threadIdx.x & 31, ty = threadIdx.x >> 5;
  const __bf16* src = qkv + (long)b * KS * QKVN + 2 * DD + h * HD;
#pragma unroll
  for (int i = ty; i < 32; i += 8) t[i][tx] = src[(long)(kk0 + i) * QKVN + n0 + tx];
  __syncthreads();
  __bf16* dst = vt + (long)z * HD * KS;
#pragma unroll
  for (int i = ty; i < 32; i += 8) dst[(long)(n0 + i) * KS + kk0 + tx] = t[tx][i];
}

// ---------------------------------------------------------------------------
// Gather selected rows, keep fp32 copy for residual, RMSNorm -> bf16.
// ---------------------------------------------------------------------------
__global__ __launch_bounds__(256) void k_gather_norm(const float* __restrict__ x,
                                                     const int* __restrict__ idx,
                                                     const float* __restrict__ w,
                                                     float* __restrict__ xsel,
                                                     __bf16* __restrict__ h1) {
  int r = blockIdx.x;
  int b = r >> 9;
  int tok = idx[r];
  const float* src = x + ((long)b * TT + tok) * DD;
  float* xd = xsel + (long)r * DD;
  __bf16* hd = h1 + (long)r * DD;
  float vals[8];
  float ss = 0.f;
#pragma unroll
  for (int i = 0; i < 8; ++i) {
    int c = threadIdx.x + i * 256;
    float v = src[c];
    vals[i] = v;
    ss += v * v;
    xd[c] = v;
  }
  __shared__ float red[256];
  red[threadIdx.x] = ss;
  __syncthreads();
  for (int o = 128; o > 0; o >>= 1) {
    if (threadIdx.x < o) red[threadIdx.x] += red[threadIdx.x + o];
    __syncthreads();
  }
  float rstd = rsqrtf(red[0] / (float)DD + EPSF);
#pragma unroll
  for (int i = 0; i < 8; ++i) {
    int c = threadIdx.x + i * 256;
    hd[c] = (__bf16)(vals[i] * rstd * w[c]);
  }
}

// ---------------------------------------------------------------------------
// Batched bf16 WMMA GEMM, async double-buffered LDS staging.
//   C[z] = scale * A[z] (M x Kd row-major bf16) x B[z]^T
//   where B[z] is stored N-major: (N x Kd row-major bf16).
// Per-z offset = (z/zdiv)*outer + (z%zdiv)*inner for A, B, C.
// Optional causal mask on original positions (attention-score epilogue).
// Block: 256 threads (8 wave32), 128x128 C tile, K-step 32.
// ---------------------------------------------------------------------------
__global__ __launch_bounds__(256) void k_gemm_bf16(
    const __bf16* __restrict__ A, long a_base, int lda, long a_outer, long a_inner,
    const __bf16* __restrict__ Bm, long b_base, int ldb, long b_outer, long b_inner,
    float* __restrict__ Cf, __bf16* __restrict__ Cb, long c_base, int ldc,
    long c_outer, long c_inner,
    int zdiv, int Kd, float scale,
    const int* __restrict__ pos, int do_mask) {
  __shared__ __bf16 Alds[2][128][40];
  __shared__ __bf16 Blds[2][128][40];

  int z = blockIdx.z;
  int zq = z / zdiv, zr = z - zq * zdiv;
  const __bf16* Ab = A + a_base + (long)zq * a_outer + (long)zr * a_inner;
  const __bf16* Bb = Bm + b_base + (long)zq * b_outer + (long)zr * b_inner;

  int tileM = blockIdx.y << 7, tileN = blockIdx.x << 7;
  int tid = threadIdx.x;
  int lane = tid & 31, laneM = lane & 15, half = lane >> 4;
  int w = tid >> 5, wr = w >> 2, wc = w & 3;
  int rowBase = wr << 6, colBase = wc << 5;

  // Per-thread staging chunks: rows r0/r0+64, 8-element column group.
  int r0 = tid >> 2, c8 = (tid & 3) << 3;
  const __bf16* ga0 = Ab + (long)(tileM + r0) * lda + c8;
  const __bf16* ga1 = Ab + (long)(tileM + r0 + 64) * lda + c8;
  const __bf16* gb0 = Bb + (long)(tileN + r0) * ldb + c8;
  const __bf16* gb1 = Bb + (long)(tileN + r0 + 64) * ldb + c8;

  v8f acc[4][2] = {};

  int nk = Kd >> 5;
  // Prologue: stream tile 0 into buffer 0.
  async_b128(&Alds[0][r0][c8], ga0);
  async_b128(&Alds[0][r0 + 64][c8], ga1);
  async_b128(&Blds[0][r0][c8], gb0);
  async_b128(&Blds[0][r0 + 64][c8], gb1);

  for (int i = 0; i < nk; ++i) {
    int buf = i & 1;
    bool more = (i + 1) < nk;
    if (more) {
      int k1 = (i + 1) << 5;
      async_b128(&Alds[buf ^ 1][r0][c8], ga0 + k1);
      async_b128(&Alds[buf ^ 1][r0 + 64][c8], ga1 + k1);
      async_b128(&Blds[buf ^ 1][r0][c8], gb0 + k1);
      async_b128(&Blds[buf ^ 1][r0 + 64][c8], gb1 + k1);
      // In-order completion: <=4 outstanding => tile i has landed.
      asm volatile("s_wait_asynccnt 0x4" ::: "memory");
    } else {
      asm volatile("s_wait_asynccnt 0x0" ::: "memory");
    }
    __syncthreads();

    v16bf afr[4], bfr[2];
#pragma unroll
    for (int tr = 0; tr < 4; ++tr)
      afr[tr] = ld_frag(&Alds[buf][rowBase + (tr << 4) + laneM][0], half);
#pragma unroll
    for (int tc = 0; tc < 2; ++tc)
      bfr[tc] = ld_frag(&Blds[buf][colBase + (tc << 4) + laneM][0], half);
#pragma unroll
    for (int tr = 0; tr < 4; ++tr)
#pragma unroll
      for (int tc = 0; tc < 2; ++tc)
        acc[tr][tc] = __builtin_amdgcn_wmma_f32_16x16x32_bf16(
            false, afr[tr], false, bfr[tc], (short)0, acc[tr][tc], false, false);
    __syncthreads();  // all waves done with buf before it is refilled
  }

  long cz = c_base + (long)zq * c_outer + (long)zr * c_inner;
#pragma unroll
  for (int tr = 0; tr < 4; ++tr) {
#pragma unroll
    for (int tc = 0; tc < 2; ++tc) {
      int n = tileN + colBase + (tc << 4) + laneM;
#pragma unroll
      for (int i = 0; i < 8; ++i) {
        int m = tileM + rowBase + (tr << 4) + i + (half << 3);
        float v = acc[tr][tc][i] * scale;
        if (do_mask) {
          const int* pb = pos + (long)zq * KS;
          if (pb[m] < pb[n]) v = -__builtin_inff();
        }
        long cidx = cz + (long)m * ldc + n;
        if (Cb) Cb[cidx] = (__bf16)v;
        else Cf[cidx] = v;
      }
    }
  }
}

// ---------------------------------------------------------------------------
// Row softmax over 512 columns (masked entries are -inf -> weight 0).
// ---------------------------------------------------------------------------
__global__ __launch_bounds__(256) void k_softmax(const float* __restrict__ S,
                                                 __bf16* __restrict__ P) {
  long row = blockIdx.x;
  const float* sp = S + row * KS;
  __bf16* pp = P + row * KS;
  int tid = threadIdx.x;
  float v0 = sp[tid], v1 = sp[tid + 256];
  __shared__ float red[256];
  __shared__ float bcast;
  red[tid] = fmaxf(v0, v1);
  __syncthreads();
  for (int o = 128; o > 0; o >>= 1) {
    if (tid < o) red[tid] = fmaxf(red[tid], red[tid + o]);
    __syncthreads();
  }
  if (tid == 0) bcast = red[0];
  __syncthreads();
  float mx = bcast;
  float e0 = expf(v0 - mx), e1 = expf(v1 - mx);
  red[tid] = e0 + e1;
  __syncthreads();
  for (int o = 128; o > 0; o >>= 1) {
    if (tid < o) red[tid] += red[tid + o];
    __syncthreads();
  }
  float inv = 1.f / red[0];
  pp[tid] = (__bf16)(e0 * inv);
  pp[tid + 256] = (__bf16)(e1 * inv);
}

// ---------------------------------------------------------------------------
// xres = xsel + oproj; h2 = rmsnorm(xres) * norm2_w (bf16)
// ---------------------------------------------------------------------------
__global__ __launch_bounds__(256) void k_residual_norm(const float* __restrict__ xsel,
                                                       const float* __restrict__ oproj,
                                                       const float* __restrict__ w,
                                                       float* __restrict__ xres,
                                                       __bf16* __restrict__ h2) {
  int r = blockIdx.x;
  const float* a = xsel + (long)r * DD;
  const float* o = oproj + (long)r * DD;
  float* xd = xres + (long)r * DD;
  __bf16* hd = h2 + (long)r * DD;
  float vals[8];
  float ss = 0.f;
#pragma unroll
  for (int i = 0; i < 8; ++i) {
    int c = threadIdx.x + i * 256;
    float v = a[c] + o[c];
    vals[i] = v;
    ss += v * v;
    xd[c] = v;
  }
  __shared__ float red[256];
  red[threadIdx.x] = ss;
  __syncthreads();
  for (int of = 128; of > 0; of >>= 1) {
    if (threadIdx.x < of) red[threadIdx.x] += red[threadIdx.x + of];
    __syncthreads();
  }
  float rstd = rsqrtf(red[0] / (float)DD + EPSF);
#pragma unroll
  for (int i = 0; i < 8; ++i) {
    int c = threadIdx.x + i * 256;
    hd[c] = (__bf16)(vals[i] * rstd * w[c]);
  }
}

// ---------------------------------------------------------------------------
// g = bf16( silu(ff1) * ff2 )
// ---------------------------------------------------------------------------
__global__ __launch_bounds__(256) void k_swiglu(const float* __restrict__ f1,
                                                const float* __restrict__ f2,
                                                __bf16* __restrict__ g, long total) {
  long i = (long)blockIdx.x * 256 + threadIdx.x;
  if (i >= total) return;
  float a = f1[i];
  float s = a / (1.f + expf(-a));
  g[i] = (__bf16)(s * f2[i]);
}

// ---------------------------------------------------------------------------
// out[b, idx[r], :] = xres[r, :] + ffout[r, :]
// ---------------------------------------------------------------------------
__global__ __launch_bounds__(256) void k_scatter(const float* __restrict__ xres,
                                                 const float* __restrict__ ffout,
                                                 const int* __restrict__ idx,
                                                 float* __restrict__ out) {
  int r = blockIdx.x;
  int b = r >> 9;
  int tok = idx[r];
  float* dst = out + ((long)b * TT + tok) * DD;
  const float* a = xres + (long)r * DD;
  const float* f = ffout + (long)r * DD;
  for (int c = threadIdx.x; c < DD; c += 256) dst[c] = a[c] + f[c];
}

// ---------------------------------------------------------------------------
extern "C" void kernel_launch(void* const* d_in, const int* in_sizes, int n_in,
                              void* d_out, int out_size, void* d_ws, size_t ws_size,
                              hipStream_t stream) {
  (void)in_sizes; (void)n_in; (void)out_size; (void)ws_size;
  const float* x            = (const float*)d_in[0];
  const int*   position_ids = (const int*)d_in[1];
  const float* router_w     = (const float*)d_in[2];
  const float* norm1_w      = (const float*)d_in[3];
  const float* qkv_w        = (const float*)d_in[4];
  const float* out_w        = (const float*)d_in[5];
  const float* norm2_w      = (const float*)d_in[6];
  const float* w1           = (const float*)d_in[7];
  const float* w2           = (const float*)d_in[8];
  const float* w3           = (const float*)d_in[9];
  float* out = (float*)d_out;
  char* ws = (char*)d_ws;

  // ---- workspace arena (all sizes multiples of 256) ----
  size_t off = 0;
  auto alloc = [&](size_t bytes) {
    size_t o = off;
    off += (bytes + 255) & ~(size_t)255;
    return o;
  };
  size_t o_scores = alloc((size_t)BB * TT * 4);
  size_t o_idx    = alloc((size_t)MM * 4);
  size_t o_pos    = alloc((size_t)MM * 4);
  size_t o_xsel   = alloc((size_t)MM * DD * 4);
  size_t o_h1     = alloc((size_t)MM * DD * 2);           // reused as h2
  size_t o_qkv    = alloc((size_t)MM * QKVN * 2);         // + o_wqT reused as ff2 (fp32)
  size_t o_wqT    = alloc((size_t)QKVN * DD * 2);         // qkv_w^T  (6144 x 2048)
  size_t o_woutT  = alloc((size_t)DD * DD * 2);           // out_w^T  (2048 x 2048)
  size_t o_w1T    = alloc((size_t)DFFP * DD * 2);         // w1^T (5504 x 2048); reused as g
  size_t o_w2T    = alloc((size_t)DFFP * DD * 2);         // w2^T (5504 x 2048)
  size_t o_w3T    = alloc((size_t)DD * DFFP * 2);         // w3^T (2048 x 5504)
  size_t o_vt     = alloc((size_t)BB * HH * HD * KS * 2); // V^T per (b,h): 128 x 512
  size_t o_S      = alloc((size_t)BB * HH * KS * KS * 4); // reused as ff1 (fp32)
  size_t o_P      = alloc((size_t)BB * HH * KS * KS * 2);
  size_t o_oasm   = alloc((size_t)MM * DD * 2);
  size_t o_oproj  = alloc((size_t)MM * DD * 4);           // reused as ffout
  size_t o_xres   = alloc((size_t)MM * DD * 4);
  // aliases (time-disjoint)
  size_t o_h2 = o_h1, o_ff1 = o_S, o_ff2 = o_qkv, o_g = o_w1T, o_ffout = o_oproj;

#define BF16P(o) ((__bf16*)(ws + (o)))
#define F32P(o)  ((float*)(ws + (o)))
#define I32P(o)  ((int*)(ws + (o)))

  // Pass-through tokens: out = x
  hipMemcpyAsync(out, x, (size_t)BB * TT * DD * sizeof(float),
                 hipMemcpyDeviceToDevice, stream);

  // Router + top-K
  k_router<<<BB * TT, 256, 0, stream>>>(x, router_w, F32P(o_scores));
  k_topk<<<BB, 1024, 0, stream>>>(F32P(o_scores), position_ids, I32P(o_idx), I32P(o_pos));

  // Weight conversions: transposed bf16, zero-padded for D_FF.
  k_cvt_padT<<<dim3(DD / 32, QKVN / 32), 256, 0, stream>>>(qkv_w, BF16P(o_wqT),
                                                           DD, QKVN, QKVN, DD);
  k_cvt_padT<<<dim3(DD / 32, DD / 32), 256, 0, stream>>>(out_w, BF16P(o_woutT),
                                                         DD, DD, DD, DD);
  k_cvt_padT<<<dim3(DD / 32, DFFP / 32), 256, 0, stream>>>(w1, BF16P(o_w1T),
                                                           DD, DFF, DFFP, DD);
  k_cvt_padT<<<dim3(DD / 32, DFFP / 32), 256, 0, stream>>>(w2, BF16P(o_w2T),
                                                           DD, DFF, DFFP, DD);
  k_cvt_padT<<<dim3(DFFP / 32, DD / 32), 256, 0, stream>>>(w3, BF16P(o_w3T),
                                                           DFF, DD, DD, DFFP);

  // Gather + RMSNorm1
  k_gather_norm<<<MM, 256, 0, stream>>>(x, I32P(o_idx), norm1_w, F32P(o_xsel), BF16P(o_h1));

  // GEMM1: qkv(bf16) = h1 (2048x2048) x qkv_w; B = wqT (6144 x 2048 N-major)
  k_gemm_bf16<<<dim3(QKVN / 128, MM / 128, 1), 256, 0, stream>>>(
      BF16P(o_h1), 0, DD, 0, 0,
      BF16P(o_wqT), 0, DD, 0, 0,
      nullptr, BF16P(o_qkv), 0, QKVN, 0, 0,
      1, DD, 1.f, nullptr, 0);

  // V^T for the PV GEMM
  k_transpose_v<<<dim3(KS / 32, HD / 32, BB * HH), 256, 0, stream>>>(
      BF16P(o_qkv), BF16P(o_vt));

  // Attention scores: S[z] = scale * Q K^T, z = b*H + h, causal mask on pos
  // B = K rows (N-major already inside qkv).
  k_gemm_bf16<<<dim3(KS / 128, KS / 128, BB * HH), 256, 0, stream>>>(
      BF16P(o_qkv), 0, QKVN, (long)KS * QKVN, HD,
      BF16P(o_qkv), DD, QKVN, (long)KS * QKVN, HD,
      F32P(o_S), nullptr, 0, KS, (long)HH * KS * KS, (long)KS * KS,
      HH, HD, 0.08838834764831845f, I32P(o_pos), 1);

  // Softmax rows
  k_softmax<<<BB * HH * KS, 256, 0, stream>>>(F32P(o_S), BF16P(o_P));

  // O = P V, assembled into (2048 x 2048) bf16 at column h*128; B = Vt.
  k_gemm_bf16<<<dim3(HD / 128, KS / 128, BB * HH), 256, 0, stream>>>(
      BF16P(o_P), 0, KS, (long)HH * KS * KS, (long)KS * KS,
      BF16P(o_vt), 0, KS, (long)HH * HD * KS, (long)HD * KS,
      nullptr, BF16P(o_oasm), 0, DD, (long)KS * DD, HD,
      HH, KS, 1.f, nullptr, 0);

  // GEMM2: oproj(f32) = o_asm x out_w; B = woutT
  k_gemm_bf16<<<dim3(DD / 128, MM / 128, 1), 256, 0, stream>>>(
      BF16P(o_oasm), 0, DD, 0, 0,
      BF16P(o_woutT), 0, DD, 0, 0,
      F32P(o_oproj), nullptr, 0, DD, 0, 0,
      1, DD, 1.f, nullptr, 0);

  // Residual + RMSNorm2
  k_residual_norm<<<MM, 256, 0, stream>>>(F32P(o_xsel), F32P(o_oproj), norm2_w,
                                          F32P(o_xres), BF16P(o_h2));

  // GEMM3a/b: ff1 = h2 x w1, ff2 = h2 x w2  (2048 x 5504); B = w1T/w2T
  k_gemm_bf16<<<dim3(DFFP / 128, MM / 128, 1), 256, 0, stream>>>(
      BF16P(o_h2), 0, DD, 0, 0,
      BF16P(o_w1T), 0, DD, 0, 0,
      F32P(o_ff1), nullptr, 0, DFFP, 0, 0,
      1, DD, 1.f, nullptr, 0);
  k_gemm_bf16<<<dim3(DFFP / 128, MM / 128, 1), 256, 0, stream>>>(
      BF16P(o_h2), 0, DD, 0, 0,
      BF16P(o_w2T), 0, DD, 0, 0,
      F32P(o_ff2), nullptr, 0, DFFP, 0, 0,
      1, DD, 1.f, nullptr, 0);

  // SwiGLU (g overwrites w1T region -- w1T is dead by now)
  {
    long t = (long)MM * DFFP;
    k_swiglu<<<(t + 255) / 256, 256, 0, stream>>>(F32P(o_ff1), F32P(o_ff2), BF16P(o_g), t);
  }

  // GEMM4: ffout = g (2048x5504) x w3; B = w3T (2048 x 5504 N-major)
  k_gemm_bf16<<<dim3(DD / 128, MM / 128, 1), 256, 0, stream>>>(
      BF16P(o_g), 0, DFFP, 0, 0,
      BF16P(o_w3T), 0, DFFP, 0, 0,
      F32P(o_ffout), nullptr, 0, DD, 0, 0,
      1, DFFP, 1.f, nullptr, 0);

  // Final residual + scatter
  k_scatter<<<MM, 256, 0, stream>>>(F32P(o_xres), F32P(o_ffout), I32P(o_idx), out);

#undef BF16P
#undef F32P
#undef I32P
}
